// CornerPool_module_11106785428027
// MI455X (gfx1250) — matrix operand order repaired and checked by reading, not compiled, not run on hardware
//
#include <hip/hip_runtime.h>

typedef __bf16 bf16_t;
typedef __attribute__((ext_vector_type(16))) __bf16 v16bf;
typedef __attribute__((ext_vector_type(8)))  __bf16 v8bf;
typedef __attribute__((ext_vector_type(8)))  float  v8f;

// ---------------------------------------------------------------------------
// gfx1250 async copy: 16 bytes global -> LDS per lane, tracked by ASYNCcnt.
// Generic pointers to __shared__ carry the LDS byte offset in their low 32
// bits (LDS aperture mapping, ISA 10.2), which is exactly the VDST operand.
// ---------------------------------------------------------------------------
__device__ __forceinline__ void async_copy_b128(void* lds_ptr, const void* gptr) {
    unsigned int lds_off = (unsigned int)(unsigned long long)lds_ptr;
    asm volatile("global_load_async_to_lds_b128 %0, %1, off"
                 :: "v"(lds_off), "v"(gptr)
                 : "memory");
}
__device__ __forceinline__ void wait_asynccnt0() {
    asm volatile("s_wait_asynccnt 0" ::: "memory");
}

// ---------------------------------------------------------------------------
// fp32 -> bf16 conversion
// ---------------------------------------------------------------------------
__global__ void cvt_f32_bf16(const float* __restrict__ in, bf16_t* __restrict__ out, int n) {
    int i = blockIdx.x * 256 + threadIdx.x;
    if (i < n) out[i] = (bf16_t)in[i];
}

// weight repack: [Cout][Cin][pos] fp32  ->  [pos][Cout][Cin] bf16 (Cin contiguous)
__global__ void repack_w_bf16(const float* __restrict__ in, bf16_t* __restrict__ out,
                              int Cout, int Cin, int npos) {
    int i = blockIdx.x * 256 + threadIdx.x;
    int total = Cout * Cin * npos;
    if (i >= total) return;
    int ci  = i % Cin;
    int co  = (i / Cin) % Cout;
    int pos = i / (Cin * Cout);
    out[i] = (bf16_t)in[((size_t)co * Cin + ci) * npos + pos];
}

// a + b -> bf16
__global__ void add2_bf16(const float* __restrict__ a, const float* __restrict__ b,
                          bf16_t* __restrict__ out, int n) {
    int i = blockIdx.x * 256 + threadIdx.x;
    if (i < n) out[i] = (bf16_t)(a[i] + b[i]);
}

// ---------------------------------------------------------------------------
// cummax scans (in place).  Tensor is (8, C, 128, 128) fp32.
// ---------------------------------------------------------------------------
template <bool REV>
__global__ void cummax_h(float* __restrict__ t, int C) {
    int idx = blockIdx.x * 256 + threadIdx.x;      // over B*C*W columns
    int total = 8 * C * 128;
    if (idx >= total) return;
    int w = idx & 127;
    int bc = idx >> 7;
    float* p = t + (size_t)bc * 128 * 128 + w;
    if (REV) {
        float m = p[127 * 128];
        for (int h = 126; h >= 0; --h) { float v = p[h * 128]; m = fmaxf(m, v); p[h * 128] = m; }
    } else {
        float m = p[0];
        for (int h = 1; h < 128; ++h) { float v = p[h * 128]; m = fmaxf(m, v); p[h * 128] = m; }
    }
}

template <bool REV>
__global__ void cummax_w(float* __restrict__ t, int C) {
    int idx = blockIdx.x * 256 + threadIdx.x;      // over B*C*H rows
    int total = 8 * C * 128;
    if (idx >= total) return;
    float* p = t + (size_t)idx * 128;
    if (REV) {
        float m = p[127];
        for (int w = 126; w >= 0; --w) { float v = p[w]; m = fmaxf(m, v); p[w] = m; }
    } else {
        float m = p[0];
        for (int w = 1; w < 128; ++w) { float v = p[w]; m = fmaxf(m, v); p[w] = m; }
    }
}

// ---------------------------------------------------------------------------
// Implicit-GEMM conv (KSxKS, SAME pad) + BN affine epilogue via WMMA bf16.
// Block = 256 threads (8 waves).  One block computes:
//   batch bz,  TWO output rows (h0, h0+1),  32 output channels,  128 columns.
// Wave w owns columns [16w,16w+16) with FOUR 16x16 f32 accumulators
// (2 cout tiles x 2 rows) -> 4 WMMAs per tap, A reused across rows,
// B reused across cout tiles.  Staging uses async global->LDS b128 copies.
// MODE 0: out = relu(acc*s+b)            -> fp32 outf
// MODE 1: out = acc*s+b                  -> fp32 outf
// MODE 2: out = relu(acc*s+b + resid)    -> bf16 outb
// ---------------------------------------------------------------------------
template <int KS, int MODE>
__global__ __launch_bounds__(256)
void conv_wmma(const bf16_t* __restrict__ in, int Cin,
               const bf16_t* __restrict__ wgt, int Cout,   // packed [pos][Cout][Cin]
               const float* __restrict__ scale, const float* __restrict__ bias,
               float* __restrict__ outf,
               const float* __restrict__ resid,
               bf16_t* __restrict__ outb)
{
    constexpr int H = 128, W = 128;
    constexpr int NPOS = KS * KS;
    constexpr int NROW = (KS == 3) ? 4 : 2;   // slab rows (2 outputs + halo)
    constexpr int HALO = (KS == 3) ? 1 : 0;
    constexpr int SSTR = 152;   // slab row stride (bf16): 304B, 16B-aligned, low-conflict
    constexpr int WSTR = 40;    // weight row stride (bf16): 80B, 16B-aligned, low-conflict

    const int h0       = blockIdx.x * 2;
    const int coutBase = blockIdx.y * 32;
    const int bz       = blockIdx.z;
    const int tid      = threadIdx.x;
    const int wave     = tid >> 5;
    const int lane     = tid & 31;
    const int halfk    = lane >> 4;
    const int lm       = lane & 15;

    __shared__ __align__(16) bf16_t sIn[NROW * 32 * SSTR];    // [row][cin(32)][col]
    __shared__ __align__(16) bf16_t sW [NPOS * 32 * WSTR];    // [pos][m(32)][k(32)+pad]

    v8f c00 = {}, c01 = {}, c10 = {}, c11 = {};   // c[mtile][row]

    // Halo columns (gcol==-1 / gcol==128) are always zero; write them once.
    if (KS == 3) {
        int row = tid >> 6, rem = tid & 63, cin = rem >> 1, side = rem & 1;
        sIn[(row * 32 + cin) * SSTR + (side ? (8 + W) : 7)] = (bf16_t)0.0f;
    }

    for (int cinBase = 0; cinBase < Cin; cinBase += 32) {
        // ---- stage input interior: async 16B global->LDS copies ----
        constexpr int NV = NROW * 32 * (W / 8);
        for (int idx = tid; idx < NV; idx += 256) {
            int row = idx / (32 * 16);
            int rem = idx % (32 * 16);
            int cin = rem >> 4;
            int v8  = rem & 15;
            int hsrc = h0 - HALO + row;
            bf16_t* lp = &sIn[(row * 32 + cin) * SSTR + 8 + v8 * 8];
            if (hsrc >= 0 && hsrc < H) {
                const bf16_t* gp = &in[(((size_t)bz * Cin + cinBase + cin) * H + hsrc) * W + v8 * 8];
                async_copy_b128(lp, gp);
                if (cinBase + 32 < Cin) __builtin_prefetch(gp + 32 * H * W, 0, 1);
            } else {
                v8bf z = {};
                *(v8bf*)lp = z;
            }
        }
        // ---- stage weights (repacked layout: Cin contiguous) ----
        constexpr int NWV = NPOS * 32 * 4;   // 4 vectors of 8 per (pos,m)
        for (int idx = tid; idx < NWV; idx += 256) {
            int pos = idx >> 7;
            int rem = idx & 127;
            int m  = rem >> 2;
            int k8 = rem & 3;
            const bf16_t* gp = &wgt[((size_t)pos * Cout + coutBase + m) * Cin + cinBase + k8 * 8];
            async_copy_b128(&sW[pos * (32 * WSTR) + m * WSTR + k8 * 8], gp);
        }
        wait_asynccnt0();    // this wave's async copies have landed in LDS
        __syncthreads();     // cross-wave visibility

        #pragma unroll
        for (int pos = 0; pos < NPOS; ++pos) {
            const int ky = pos / KS;
            const int dx = (KS == 3) ? (pos % KS) : 0;

            // A fragments (M x K = 16x32): M = lane%16, interleaved K pairs per half
            v16bf a0, a1;
            const bf16_t* ar0 = &sW[pos * (32 * WSTR) + lm * WSTR];
            const bf16_t* ar1 = &sW[pos * (32 * WSTR) + (16 + lm) * WSTR];
            #pragma unroll
            for (int e = 0; e < 16; ++e) {
                int v = e >> 1, j = e & 1;
                int K = ((v < 4) ? 2 * v : 16 + 2 * (v - 4)) + halfk * 8 + j;
                a0[e] = ar0[K];
                a1[e] = ar1[K];
            }

            // B fragments (K x N = 32x16): N = lane%16, K interleaved per half (same order as A)
            #pragma unroll
            for (int r = 0; r < 2; ++r) {
                const bf16_t* br = &sIn[((r + ky) * 32) * SSTR + 8 - HALO + dx + wave * 16 + lm];
                v16bf bv;
                #pragma unroll
                for (int e = 0; e < 16; ++e) bv[e] = br[(((e < 8) ? e : 8 + e) + halfk * 8) * SSTR];

                if (r == 0) {
                    c00 = __builtin_amdgcn_wmma_f32_16x16x32_bf16(false, a0, false, bv, (short)0, c00, false, false);
                    c10 = __builtin_amdgcn_wmma_f32_16x16x32_bf16(false, a1, false, bv, (short)0, c10, false, false);
                } else {
                    c01 = __builtin_amdgcn_wmma_f32_16x16x32_bf16(false, a0, false, bv, (short)0, c01, false, false);
                    c11 = __builtin_amdgcn_wmma_f32_16x16x32_bf16(false, a1, false, bv, (short)0, c11, false, false);
                }
            }
        }
        __syncthreads();
    }

    // ---- epilogue: D layout lane L -> N=L%16, VGPR r -> M = r + 8*(L/16) ----
    const int col = wave * 16 + lm;
    auto emit = [&](int co, int hh, float v) {
        size_t o = (((size_t)bz * Cout + co) * H + hh) * W + col;
        if (MODE == 2) { v += resid[o]; v = fmaxf(v, 0.0f); outb[o] = (bf16_t)v; }
        else           { if (MODE == 0) v = fmaxf(v, 0.0f); outf[o] = v; }
    };
    #pragma unroll
    for (int rr = 0; rr < 8; ++rr) {
        int m0  = rr + 8 * halfk;
        int coA = coutBase + m0;
        int coB = coutBase + 16 + m0;
        float sA = scale[coA], bA = bias[coA];
        float sB = scale[coB], bB = bias[coB];
        emit(coA, h0,     c00[rr] * sA + bA);
        emit(coA, h0 + 1, c01[rr] * sA + bA);
        emit(coB, h0,     c10[rr] * sB + bB);
        emit(coB, h0 + 1, c11[rr] * sB + bB);
    }
}

// ---------------------------------------------------------------------------
extern "C" void kernel_launch(void* const* d_in, const int* in_sizes, int n_in,
                              void* d_out, int out_size, void* d_ws, size_t ws_size,
                              hipStream_t stream) {
    (void)in_sizes; (void)n_in; (void)out_size; (void)ws_size;

    const float* x = (const float*)d_in[0];
    // param order: t l b r tl3 br3 tl1 br1 tlo bro ; each (w, s, b)
    const float* w_t   = (const float*)d_in[1];  const float* s_t   = (const float*)d_in[2];  const float* b_t   = (const float*)d_in[3];
    const float* w_l   = (const float*)d_in[4];  const float* s_l   = (const float*)d_in[5];  const float* b_l   = (const float*)d_in[6];
    const float* w_b   = (const float*)d_in[7];  const float* s_b   = (const float*)d_in[8];  const float* b_b   = (const float*)d_in[9];
    const float* w_r   = (const float*)d_in[10]; const float* s_r   = (const float*)d_in[11]; const float* b_r   = (const float*)d_in[12];
    const float* w_tl3 = (const float*)d_in[13]; const float* s_tl3 = (const float*)d_in[14]; const float* b_tl3 = (const float*)d_in[15];
    const float* w_br3 = (const float*)d_in[16]; const float* s_br3 = (const float*)d_in[17]; const float* b_br3 = (const float*)d_in[18];
    const float* w_tl1 = (const float*)d_in[19]; const float* s_tl1 = (const float*)d_in[20]; const float* b_tl1 = (const float*)d_in[21];
    const float* w_br1 = (const float*)d_in[22]; const float* s_br1 = (const float*)d_in[23]; const float* b_br1 = (const float*)d_in[24];
    const float* w_tlo = (const float*)d_in[25]; const float* s_tlo = (const float*)d_in[26]; const float* b_tlo = (const float*)d_in[27];
    const float* w_bro = (const float*)d_in[28]; const float* s_bro = (const float*)d_in[29]; const float* b_bro = (const float*)d_in[30];

    const int N256 = 8 * 256 * 128 * 128;   // 33,554,432
    const int N128 = 8 * 128 * 128 * 128;   // 16,777,216
    const int WN33 = 128 * 256 * 9;         // 294,912 (3x3 128<->256)
    const int WN11 = 256 * 256;             // 65,536
    const int WNO3 = 256 * 256 * 9;         // 589,824

    // ---- workspace layout ----
    char* ws = (char*)d_ws;
    size_t off = 0;
    auto alloc = [&](size_t bytes) -> void* {
        void* p = ws + off;
        off = (off + bytes + 255) & ~(size_t)255;
        return p;
    };
    bf16_t* xb     = (bf16_t*)alloc((size_t)N256 * 2);
    bf16_t* wb_t   = (bf16_t*)alloc((size_t)WN33 * 2);
    bf16_t* wb_l   = (bf16_t*)alloc((size_t)WN33 * 2);
    bf16_t* wb_b   = (bf16_t*)alloc((size_t)WN33 * 2);
    bf16_t* wb_r   = (bf16_t*)alloc((size_t)WN33 * 2);
    bf16_t* wb_tl3 = (bf16_t*)alloc((size_t)WN33 * 2);
    bf16_t* wb_br3 = (bf16_t*)alloc((size_t)WN33 * 2);
    bf16_t* wb_tl1 = (bf16_t*)alloc((size_t)WN11 * 2);
    bf16_t* wb_br1 = (bf16_t*)alloc((size_t)WN11 * 2);
    bf16_t* wb_tlo = (bf16_t*)alloc((size_t)WNO3 * 2);
    bf16_t* wb_bro = (bf16_t*)alloc((size_t)WNO3 * 2);
    float*  bufA   = (float*) alloc((size_t)N128 * 4);   // t / b  (pooled in place)
    float*  bufB   = (float*) alloc((size_t)N128 * 4);   // l / r  (pooled in place)
    bf16_t* sum    = (bf16_t*)alloc((size_t)N128 * 2);   // tp+lp / bp+rp in bf16
    float*  mid    = (float*) alloc((size_t)N256 * 4);   // tl3/br3 conv out (pre-merge)
    bf16_t* midb   = (bf16_t*)alloc((size_t)N256 * 2);   // merged+relu in bf16

    float* outTL = (float*)d_out;
    float* outBR = outTL + (size_t)N256;

    auto repack = [&](const float* src, bf16_t* dst, int Cout, int Cin, int npos) {
        int n = Cout * Cin * npos;
        repack_w_bf16<<<(n + 255) / 256, 256, 0, stream>>>(src, dst, Cout, Cin, npos);
    };

    // ---- precision staging ----
    cvt_f32_bf16<<<(N256 + 255) / 256, 256, 0, stream>>>(x, xb, N256);
    repack(w_t,   wb_t,   128, 256, 9);  repack(w_l,   wb_l,   128, 256, 9);
    repack(w_b,   wb_b,   128, 256, 9);  repack(w_r,   wb_r,   128, 256, 9);
    repack(w_tl3, wb_tl3, 256, 128, 9);  repack(w_br3, wb_br3, 256, 128, 9);
    repack(w_tl1, wb_tl1, 256, 256, 1);  repack(w_br1, wb_br1, 256, 256, 1);
    repack(w_tlo, wb_tlo, 256, 256, 9);  repack(w_bro, wb_bro, 256, 256, 9);

    const dim3 blk(256);
    const dim3 g128(64, 4, 8);   // Cout=128, 2 rows/block
    const dim3 g256(64, 8, 8);   // Cout=256, 2 rows/block
    const int scanBlocks = (8 * 128 * 128 + 255) / 256;

    // ================= TL branch =================
    conv_wmma<3, 0><<<g128, blk, 0, stream>>>(xb, 256, wb_t, 128, s_t, b_t, bufA, nullptr, nullptr);
    conv_wmma<3, 0><<<g128, blk, 0, stream>>>(xb, 256, wb_l, 128, s_l, b_l, bufB, nullptr, nullptr);
    cummax_h<true ><<<scanBlocks, 256, 0, stream>>>(bufA, 128);   // top pool (reverse H)
    cummax_w<true ><<<scanBlocks, 256, 0, stream>>>(bufB, 128);   // left pool (reverse W)
    add2_bf16<<<(N128 + 255) / 256, 256, 0, stream>>>(bufA, bufB, sum, N128);
    conv_wmma<3, 1><<<g256, blk, 0, stream>>>(sum, 128, wb_tl3, 256, s_tl3, b_tl3, mid, nullptr, nullptr);
    conv_wmma<1, 2><<<g256, blk, 0, stream>>>(xb, 256, wb_tl1, 256, s_tl1, b_tl1, nullptr, mid, midb);
    conv_wmma<3, 0><<<g256, blk, 0, stream>>>(midb, 256, wb_tlo, 256, s_tlo, b_tlo, outTL, nullptr, nullptr);

    // ================= BR branch =================
    conv_wmma<3, 0><<<g128, blk, 0, stream>>>(xb, 256, wb_b, 128, s_b, b_b, bufA, nullptr, nullptr);
    conv_wmma<3, 0><<<g128, blk, 0, stream>>>(xb, 256, wb_r, 128, s_r, b_r, bufB, nullptr, nullptr);
    cummax_h<false><<<scanBlocks, 256, 0, stream>>>(bufA, 128);   // bottom pool (forward H)
    cummax_w<false><<<scanBlocks, 256, 0, stream>>>(bufB, 128);   // right pool (forward W)
    add2_bf16<<<(N128 + 255) / 256, 256, 0, stream>>>(bufA, bufB, sum, N128);
    conv_wmma<3, 1><<<g256, blk, 0, stream>>>(sum, 128, wb_br3, 256, s_br3, b_br3, mid, nullptr, nullptr);
    conv_wmma<1, 2><<<g256, blk, 0, stream>>>(xb, 256, wb_br1, 256, s_br1, b_br1, nullptr, mid, midb);
    conv_wmma<3, 0><<<g256, blk, 0, stream>>>(midb, 256, wb_bro, 256, s_bro, b_bro, outBR, nullptr, nullptr);
}
